// OmegaRelGraphConv_42064909697830
// MI455X (gfx1250) — compile-verified, matrix operand order checked by reading, not run on hardware
//
#include <hip/hip_runtime.h>

// ---------------------------------------------------------------------------
// OmegaRelGraphConv for MI455X (gfx1250, wave32, WMMA)
//
// msg      = (h[src] + edge_feats) @ W1[l]          (640000 x 128 x 128)
// neigh    = segment_mean(msg, dst)                  (f32 atomics into L2)
// self_msg = iso ? h @ W3[l] : h @ W2[l]
// h        = rrelu(neigh + self_msg)
//
// Matrix path: V_WMMA_F32_16X16X32_F16 (f16 A/B, f32 accumulate).
// Weights converted once to f16, stored transposed (WT[n][k]) so each lane's
// B-fragment is two contiguous 16B runs of its output column.
//
// Edge kernel is register-blocked 2x on M (32 edges per wave): every
// B-fragment pulled from LDS feeds two WMMAs, keeping B traffic at
// ~512B/WMMA, below the ~256B/clk WGP LDS ceiling.
// ---------------------------------------------------------------------------

typedef _Float16 v16h __attribute__((ext_vector_type(16)));
typedef _Float16 v8h  __attribute__((ext_vector_type(8)));
typedef float    v8f  __attribute__((ext_vector_type(8)));

#define N_NODES 10000
#define N_EDGES 640000
#define DIM     128
#define RRELU_SLOPE 0.22916666666666666f

union V16 { v16h v; _Float16 f[16]; };

// -------------------------------------------------------------------- utils
__global__ void zero_f32(float* __restrict__ p, int n) {
    int i = blockIdx.x * blockDim.x + threadIdx.x;
    if (i < n) p[i] = 0.0f;
}

__global__ void degree_kernel(const int* __restrict__ dst,
                              float* __restrict__ deg) {
    int e = blockIdx.x * blockDim.x + threadIdx.x;
    if (e < N_EDGES) atomicAdd(&deg[dst[e]], 1.0f);
}

// out[n*DIM + k] = (f16) in[k*DIM + n] ; grid.y selects (matrix, layer)
__global__ void convert_w(const float* __restrict__ W1,
                          const float* __restrict__ W2,
                          const float* __restrict__ W3,
                          _Float16* __restrict__ w1t,
                          _Float16* __restrict__ w2t,
                          _Float16* __restrict__ w3t) {
    int id  = blockIdx.x * blockDim.x + threadIdx.x;   // 0..16383
    int sel = blockIdx.y;                              // 0..5
    int l   = sel & 1;
    const float* in; _Float16* out;
    switch (sel >> 1) {
        case 0:  in = W1; out = w1t; break;
        case 1:  in = W2; out = w2t; break;
        default: in = W3; out = w3t; break;
    }
    in  += l * DIM * DIM;
    out += l * DIM * DIM;
    int k = id >> 7, n = id & 127;
    out[n * DIM + k] = (_Float16)in[id];
}

// A-fragment (16x32 f16) wave32 layout:
// lane<16 : row = lane,    K chunks {base..base+7} and {base+16..base+23}
// lane>=16: row = lane-16, complementary chunks (base includes +8)
// Across the wave the 4 fragments cover 16 rows x 128 cols exactly once.

// ---------------------------------------------------------- edge message GEMM
__launch_bounds__(256)
__global__ void edge_kernel(const float* __restrict__ h,
                            const float* __restrict__ edge_feats,
                            const int* __restrict__ src,
                            const int* __restrict__ dst,
                            const _Float16* __restrict__ w1t,   // [DIM][DIM] (n-major)
                            float* __restrict__ neigh) {
    __shared__ _Float16 wlds[DIM * DIM];                 // 32 KB
    {
        const uint4* g = (const uint4*)w1t;
        uint4*       s = (uint4*)wlds;
#pragma unroll
        for (int i = 0; i < 8; ++i)
            s[threadIdx.x + i * 256] = g[threadIdx.x + i * 256];
    }
    __syncthreads();

    const int wave = threadIdx.x >> 5;
    const int lane = threadIdx.x & 31;
    const int half = lane >> 4;
    const int r16  = lane & 15;
    const int eb0  = blockIdx.x * 256 + wave * 32;       // 32 edges per wave
    const int eb1  = eb0 + 16;

    // ---- build A fragments for both 16-edge tiles (k0 = 0,32,64,96)
    const int e0 = eb0 + r16;
    const int e1 = eb1 + r16;
    const float* h0 = h          + (size_t)src[e0] * DIM;
    const float* f0 = edge_feats + (size_t)e0      * DIM;
    const float* h1 = h          + (size_t)src[e1] * DIM;
    const float* f1 = edge_feats + (size_t)e1      * DIM;

    V16 a0[4], a1[4];
#pragma unroll
    for (int k = 0; k < 4; ++k) {
        const int base = 32 * k + half * 8;
#pragma unroll
        for (int j = 0; j < 8; ++j) {
            a0[k].f[j]     = (_Float16)(h0[base + j]      + f0[base + j]);
            a0[k].f[j + 8] = (_Float16)(h0[base + 16 + j] + f0[base + 16 + j]);
            a1[k].f[j]     = (_Float16)(h1[base + j]      + f1[base + j]);
            a1[k].f[j + 8] = (_Float16)(h1[base + 16 + j] + f1[base + 16 + j]);
        }
    }

    v8f acc0[8] = {};
    v8f acc1[8] = {};
#pragma unroll
    for (int t = 0; t < 8; ++t) {                        // 8 output n-tiles
        const int n = t * 16 + r16;
        const _Float16* wrow = wlds + n * DIM;
#pragma unroll
        for (int k = 0; k < 4; ++k) {                    // 4 K chunks of 32
            const int base = 32 * k + half * 8;
            v8h lo = *(const v8h*)(wrow + base);
            v8h hi = *(const v8h*)(wrow + base + 16);
            v16h b = __builtin_shufflevector(lo, hi,
                        0,1,2,3,4,5,6,7,8,9,10,11,12,13,14,15);
            // reuse the same B-fragment for both edge tiles
            acc0[t] = __builtin_amdgcn_wmma_f32_16x16x32_f16(
                        false, a0[k].v, false, b, (short)0, acc0[t], false, false);
            acc1[t] = __builtin_amdgcn_wmma_f32_16x16x32_f16(
                        false, a1[k].v, false, b, (short)0, acc1[t], false, false);
        }
    }

    // C/D layout: vgpr r holds M = r + 8*half, N = r16 + 16*t
    int dn0[8], dn1[8];
#pragma unroll
    for (int r = 0; r < 8; ++r) {
        dn0[r] = dst[eb0 + half * 8 + r];
        dn1[r] = dst[eb1 + half * 8 + r];
    }
#pragma unroll
    for (int t = 0; t < 8; ++t) {
        const int col = t * 16 + r16;
#pragma unroll
        for (int r = 0; r < 8; ++r) {
            atomicAdd(&neigh[(size_t)dn0[r] * DIM + col], acc0[t][r]);
            atomicAdd(&neigh[(size_t)dn1[r] * DIM + col], acc1[t][r]);
        }
    }
}

// ------------------------------------------------- node self-message + combine
__launch_bounds__(256)
__global__ void node_kernel(const float* __restrict__ h,
                            const float* __restrict__ neigh,
                            const float* __restrict__ deg,
                            const _Float16* __restrict__ w2t,
                            const _Float16* __restrict__ w3t,
                            float* __restrict__ out) {
    __shared__ _Float16 wlds2[DIM * DIM];                // 32 KB
    __shared__ _Float16 wlds3[DIM * DIM];                // 32 KB
    {
        const uint4* g2 = (const uint4*)w2t; uint4* s2 = (uint4*)wlds2;
        const uint4* g3 = (const uint4*)w3t; uint4* s3 = (uint4*)wlds3;
#pragma unroll
        for (int i = 0; i < 8; ++i) {
            s2[threadIdx.x + i * 256] = g2[threadIdx.x + i * 256];
            s3[threadIdx.x + i * 256] = g3[threadIdx.x + i * 256];
        }
    }
    __syncthreads();

    const int wave = threadIdx.x >> 5;
    const int lane = threadIdx.x & 31;
    const int half = lane >> 4;
    const int r16  = lane & 15;
    const int nb   = blockIdx.x * 128 + wave * 16;       // 16 nodes per wave
    if (nb >= N_NODES) return;                           // whole-wave uniform

    const float* hrow = h + (size_t)(nb + r16) * DIM;
    V16 a[4];
#pragma unroll
    for (int k = 0; k < 4; ++k) {
        const int base = 32 * k + half * 8;
#pragma unroll
        for (int j = 0; j < 8; ++j) {
            a[k].f[j]     = (_Float16)hrow[base + j];
            a[k].f[j + 8] = (_Float16)hrow[base + 16 + j];
        }
    }

    v8f acc2[8] = {};
    v8f acc3[8] = {};
#pragma unroll
    for (int t = 0; t < 8; ++t) {
        const int n = t * 16 + r16;
        const _Float16* w2r = wlds2 + n * DIM;
        const _Float16* w3r = wlds3 + n * DIM;
#pragma unroll
        for (int k = 0; k < 4; ++k) {
            const int base = 32 * k + half * 8;
            v8h lo2 = *(const v8h*)(w2r + base);
            v8h hi2 = *(const v8h*)(w2r + base + 16);
            v16h b2 = __builtin_shufflevector(lo2, hi2,
                        0,1,2,3,4,5,6,7,8,9,10,11,12,13,14,15);
            acc2[t] = __builtin_amdgcn_wmma_f32_16x16x32_f16(
                        false, a[k].v, false, b2, (short)0, acc2[t], false, false);
            v8h lo3 = *(const v8h*)(w3r + base);
            v8h hi3 = *(const v8h*)(w3r + base + 16);
            v16h b3 = __builtin_shufflevector(lo3, hi3,
                        0,1,2,3,4,5,6,7,8,9,10,11,12,13,14,15);
            acc3[t] = __builtin_amdgcn_wmma_f32_16x16x32_f16(
                        false, a[k].v, false, b3, (short)0, acc3[t], false, false);
        }
    }

    // per-lane output rows: node = nb + 8*half + r
    float dg[8];
#pragma unroll
    for (int r = 0; r < 8; ++r) dg[r] = deg[nb + half * 8 + r];

#pragma unroll
    for (int t = 0; t < 8; ++t) {
        const int col = t * 16 + r16;
#pragma unroll
        for (int r = 0; r < 8; ++r) {
            const int node = nb + half * 8 + r;
            const float d    = dg[r];
            const float self = (d == 0.0f) ? acc3[t][r] : acc2[t][r];
            const float nv   = neigh[(size_t)node * DIM + col] / fmaxf(d, 1.0f);
            const float x    = nv + self;
            out[(size_t)node * DIM + col] = (x >= 0.0f) ? x : RRELU_SLOPE * x;
        }
    }
}

// --------------------------------------------------------------------- launch
extern "C" void kernel_launch(void* const* d_in, const int* in_sizes, int n_in,
                              void* d_out, int out_size, void* d_ws, size_t ws_size,
                              hipStream_t stream) {
    const float* node_feats = (const float*)d_in[0];
    const float* edge_feats = (const float*)d_in[1];
    const int*   src        = (const int*)d_in[2];
    const int*   dst        = (const int*)d_in[3];
    const float* W1         = (const float*)d_in[4];   // (2,128,128)
    const float* W2         = (const float*)d_in[5];
    const float* W3         = (const float*)d_in[6];
    float*       out        = (float*)d_out;

    char* ws = (char*)d_ws;
    float*    h1    = (float*)(ws + 0);                 // 10000*128 f32
    float*    neigh = (float*)(ws + 5120000);           // 10000*128 f32
    float*    deg   = (float*)(ws + 10240000);          // 10000 f32
    _Float16* w1t   = (_Float16*)(ws + 10280192);       // 2*128*128 f16
    _Float16* w2t   = (_Float16*)(ws + 10280192 + 65536);
    _Float16* w3t   = (_Float16*)(ws + 10280192 + 131072);

    const int NF = N_NODES * DIM;                       // 1,280,000

    // init: deg = 0, neigh = 0; convert weights; degrees
    zero_f32<<<(N_NODES + 255) / 256, 256, 0, stream>>>(deg, N_NODES);
    zero_f32<<<(NF + 255) / 256, 256, 0, stream>>>(neigh, NF);
    convert_w<<<dim3(64, 6), 256, 0, stream>>>(W1, W2, W3, w1t, w2t, w3t);
    degree_kernel<<<(N_EDGES + 255) / 256, 256, 0, stream>>>(dst, deg);

    // layer 0: h = node_feats -> h1
    edge_kernel<<<N_EDGES / 256, 256, 0, stream>>>(node_feats, edge_feats, src,
                                                   dst, w1t, neigh);
    node_kernel<<<(N_NODES + 127) / 128, 256, 0, stream>>>(node_feats, neigh, deg,
                                                           w2t, w3t, h1);

    // layer 1: h = h1 -> d_out
    zero_f32<<<(NF + 255) / 256, 256, 0, stream>>>(neigh, NF);
    edge_kernel<<<N_EDGES / 256, 256, 0, stream>>>(h1, edge_feats, src, dst,
                                                   w1t + DIM * DIM, neigh);
    node_kernel<<<(N_NODES + 127) / 128, 256, 0, stream>>>(h1, neigh, deg,
                                                           w2t + DIM * DIM,
                                                           w3t + DIM * DIM, out);
}
